// NDCN_56324201120322
// MI455X (gfx1250) — matrix operand order repaired and checked by reading, not compiled, not run on hardware
//
#include <hip/hip_runtime.h>
#include <hip/hip_bf16.h>

typedef __attribute__((ext_vector_type(16))) __bf16 v16b;
typedef __attribute__((ext_vector_type(8)))  float v8f;
typedef __attribute__((ext_vector_type(4)))  unsigned int v4u;
typedef __attribute__((ext_vector_type(4)))  float v4f;

#define NN 8000
#define KK 8000
#define NC 128     // state width (2H)
#define BM 64      // rows per block in big GEMM

__device__ __forceinline__ unsigned short f2bf(float x) {
    unsigned u = __float_as_uint(x);
    unsigned r = u + 0x7FFFu + ((u >> 16) & 1u);   // round-to-nearest-even
    return (unsigned short)(r >> 16);
}

__device__ __forceinline__ v8f vzero8() {
    v8f z;
#pragma unroll
    for (int i = 0; i < 8; ++i) z[i] = 0.0f;
    return z;
}

struct Q2 { v4u a, b; };
__device__ __forceinline__ v16b ldfrag(const unsigned short* p0, const unsigned short* p1) {
    Q2 t;
    t.a = *(const v4u*)p0;
    t.b = *(const v4u*)p1;
    return __builtin_bit_cast(v16b, t);
}

// ---------------------------------------------------------------------------
// Convert A (fp32, 8000x8000) to bf16 (row-major, unchanged layout)
// ---------------------------------------------------------------------------
__global__ __launch_bounds__(256) void convA_kernel(const float* __restrict__ A,
                                                    unsigned short* __restrict__ Abf) {
    long long idx = ((long long)blockIdx.x * 256 + threadIdx.x) * 8;
    if (idx >= (long long)NN * KK) return;
    v4f f0 = *(const v4f*)(A + idx);
    v4f f1 = *(const v4f*)(A + idx + 4);
    unsigned w0 = (unsigned)f2bf(f0[0]) | ((unsigned)f2bf(f0[1]) << 16);
    unsigned w1 = (unsigned)f2bf(f0[2]) | ((unsigned)f2bf(f0[3]) << 16);
    unsigned w2 = (unsigned)f2bf(f1[0]) | ((unsigned)f2bf(f1[1]) << 16);
    unsigned w3 = (unsigned)f2bf(f1[2]) | ((unsigned)f2bf(f1[3]) << 16);
    v4u v = { w0, w1, w2, w3 };
    *(v4u*)(Abf + idx) = v;
}

// ---------------------------------------------------------------------------
// Convert Wt (fp32, [128 c][128 n] row-major) to transposed bf16: WtT[n][c]
// ---------------------------------------------------------------------------
__global__ __launch_bounds__(256) void convW_kernel(const float* __restrict__ Wt,
                                                    unsigned short* __restrict__ WtT) {
    int t = threadIdx.x;
#pragma unroll
    for (int i = 0; i < 64; ++i) {
        int lin = i * 256 + t;            // lin = c*128 + n
        int c = lin >> 7, n = lin & 127;
        WtT[n * 128 + c] = f2bf(Wt[lin]);
    }
}

// ---------------------------------------------------------------------------
// Embedding: s[n] = [ h1 | h2 ], x0 = y1 - y2.  One block per node, 128 thr.
// ---------------------------------------------------------------------------
__global__ __launch_bounds__(128) void emb_kernel(
    const float* __restrict__ y1, const float* __restrict__ y2,
    const float* __restrict__ Wi1a, const float* __restrict__ bi1a,
    const float* __restrict__ Wi1b, const float* __restrict__ bi1b,
    const float* __restrict__ Wi2a, const float* __restrict__ bi2a,
    const float* __restrict__ Wi2b, const float* __restrict__ bi2b,
    float* __restrict__ s, float* __restrict__ x0out) {
    __shared__ float th[128];
    __shared__ float ysh[32];
    int n = blockIdx.x;
    int t = threadIdx.x;
    if (t < 16)      ysh[t] = y1[n * 16 + t];
    else if (t < 32) ysh[t] = y2[n * 16 + (t - 16)];
    __syncthreads();
    int ch = t >> 6, h = t & 63;
    const float* Wa = ch ? Wi2a : Wi1a;
    const float* ba = ch ? bi2a : bi1a;
    float acc = ba[h];
#pragma unroll
    for (int i = 0; i < 16; ++i) acc += ysh[ch * 16 + i] * Wa[i * 64 + h];
    th[t] = tanhf(acc);
    __syncthreads();
    const float* Wb = ch ? Wi2b : Wi1b;
    const float* bb = ch ? bi2b : bi1b;
    float acc2 = bb[h];
#pragma unroll
    for (int j = 0; j < 64; ++j) acc2 += th[ch * 64 + j] * Wb[j * 64 + h];
    s[(size_t)n * NC + t] = acc2;
    if (t < 16) x0out[n * 16 + t] = ysh[t] - ysh[16 + t];
}

// ---------------------------------------------------------------------------
// Output heads: out1/out2[tick] = sigmoid(state[:, :64] @ Wo1 + bo1), etc.
// ---------------------------------------------------------------------------
__global__ __launch_bounds__(256) void out_kernel(
    const float* __restrict__ st, const float* __restrict__ Wo1, const float* __restrict__ bo1,
    const float* __restrict__ Wo2, const float* __restrict__ bo2,
    float* __restrict__ out1, float* __restrict__ out2, int tick) {
    int n = blockIdx.x * 256 + threadIdx.x;
    if (n >= NN) return;
    const float* row = st + (size_t)n * NC;
    float a = bo1[0], b = bo2[0];
#pragma unroll 8
    for (int j = 0; j < 64; ++j) {
        a += row[j] * Wo1[j];
        b += row[64 + j] * Wo2[j];
    }
    out1[tick * NN + n] = 1.0f / (1.0f + expf(-a));
    out2[tick * NN + n] = 1.0f / (1.0f + expf(-b));
}

// ---------------------------------------------------------------------------
// RK4 combine + produce transposed bf16 GEMM input uT[128][8000].
// mode 0: u = bf16(s)
// mode 1: acc = s + b*k      ; u = bf16(s + g*k)
// mode 2: acc = acc + b*k    ; u = bf16(s + g*k)
// mode 3: s   = acc + b*k    ; u = bf16(s_new)
// b = bscale*dt, g = gscale*dt, dt read from vt on device (graph-capture safe).
// ---------------------------------------------------------------------------
__global__ __launch_bounds__(256) void combine_kernel(
    const float* __restrict__ s, const float* __restrict__ kin,
    float* __restrict__ acc, float* __restrict__ sout, unsigned short* __restrict__ uT,
    const float* __restrict__ vt, int ti, int mode, float bscale, float gscale) {
    __shared__ unsigned short tile[128 * 130];   // pad 2 -> 65-dword row stride (no bank conflicts)
    int t = threadIdx.x;
    int nb = blockIdx.x * 128;
    float dt = vt[ti + 1] - vt[ti];
    float bco = bscale * dt, gco = gscale * dt;
#pragma unroll 4
    for (int i = 0; i < 64; ++i) {
        int lin = i * 256 + t;
        int nl = lin >> 7, col = lin & 127;
        int node = nb + nl;
        if (node < NN) {
            size_t gi = (size_t)node * NC + col;
            float e;
            if (mode == 0) {
                e = s[gi];
            } else if (mode == 3) {
                e = acc[gi] + bco * kin[gi];
                sout[gi] = e;
            } else {
                float sv = s[gi];
                float kv = kin[gi];
                float av = bco * kv + (mode == 1 ? sv : acc[gi]);
                acc[gi] = av;
                e = sv + gco * kv;
            }
            tile[nl * 130 + col] = f2bf(e);
        }
    }
    __syncthreads();
#pragma unroll 4
    for (int i = 0; i < 64; ++i) {
        int lin = i * 256 + t;
        int c = lin >> 7, nl = lin & 127;
        int node = nb + nl;
        if (node < NN)
            uT[(size_t)c * KK + node] = tile[nl * 130 + c];
    }
}

// ---------------------------------------------------------------------------
// Fused f(s): k = relu( (A @ u) @ Wt + bt )
//   A:   bf16 [8000][8000] row-major (L2-resident, 128 MB)
//   uT:  bf16 [128 n][8000 k]  (transposed so B-frags are contiguous b128 loads)
//   WtT: bf16 [128 n][128 c]   (transposed, cached in LDS)
// 256 threads = 8 waves; block tile = 64 rows x 128 cols; 125 blocks.
// Main loop: double-buffered GLOBAL_LOAD_ASYNC_TO_LDS_B128 (ASYNCcnt-tracked,
// no VGPR transit), 3 async b128 per thread per K-chunk, s_wait_asynccnt to
// pipeline chunk c+1's DMA under chunk c's 4 WMMAs.
// Epilogue: C tile -> bf16 in LDS -> 16 more WMMAs against Wt -> relu+bias.
// ---------------------------------------------------------------------------
#define BUFELT (64 * 32 + 128 * 32)   // 6144 ushort per buffer (As 64x32 | Bst 128x32)

__global__ __launch_bounds__(256) void f_gemm(
    const unsigned short* __restrict__ Abf, const unsigned short* __restrict__ uT,
    const unsigned short* __restrict__ WtT, const float* __restrict__ bt,
    float* __restrict__ kout) {
    alignas(16) __shared__ unsigned short sWt[128 * 128];     // 32 KB
    alignas(16) __shared__ unsigned short sBuf[2 * BUFELT];   // 24 KB, double-buffered tiles
                                                              // (epilogue Cs 64x128 aliases sBuf)
    const int tid = threadIdx.x;
    const int lane = tid & 31;
    const int hf = lane >> 4;          // half-wave select
    const int m16 = lane & 15;
    const int wave = tid >> 5;
    const int rstrip = wave & 3;       // 16-row strip within 64-row block tile
    const int chalf = wave >> 2;       // which 64-column half
    const int rowBase = blockIdx.x * BM;

    // Stage Wt^T into LDS (2048 x b128)
    {
        const v4u* src = (const v4u*)WtT;
        v4u* dst = (v4u*)sWt;
#pragma unroll
        for (int i = 0; i < 8; ++i) dst[tid + i * 256] = src[tid + i * 256];
    }

    // Per-thread async-copy source/dest precompute (A: 1 chunk, B: 2 chunks)
    const unsigned ldsBase = (unsigned)(uintptr_t)&sBuf[0];
    const int ar = tid >> 2, aq = tid & 3;                 // A tile: 256 b128 chunks
    const unsigned short* aSrc0 = Abf + (size_t)(rowBase + ar) * KK + aq * 8;
    const int bn0 = tid >> 1, bq0 = (tid & 1) * 2;         // B tile: 512 chunks -> 2/thread,
    const unsigned short* bSrc0 = uT + (size_t)bn0 * KK + bq0 * 8;  // adjacent pair per thread

    auto prefetch = [&](int buf, int kk) {
        unsigned asB = ldsBase + (unsigned)buf * (BUFELT * 2);
        unsigned bsB = asB + 64 * 32 * 2;
        unsigned lA = asB + (unsigned)tid * 16;
        const unsigned short* gA = aSrc0 + kk;
        asm volatile("global_load_async_to_lds_b128 %0, %1, off"
                     :: "v"(lA), "v"(gA) : "memory");
        unsigned lB = bsB + (unsigned)(bn0 * 4 + bq0) * 16;
        const unsigned short* gB = bSrc0 + kk;
        asm volatile("global_load_async_to_lds_b128 %0, %1, off"
                     :: "v"(lB), "v"(gB) : "memory");
        asm volatile("global_load_async_to_lds_b128 %0, %1, off offset:16"
                     :: "v"(lB), "v"(gB + 8) : "memory");
    };

    v8f cacc[4];
#pragma unroll
    for (int i = 0; i < 4; ++i) cacc[i] = vzero8();

    prefetch(0, 0);
    int cur = 0;
    for (int kk = 0; kk < KK; kk += 32) {
        if (kk + 32 < KK) {
            prefetch(cur ^ 1, kk + 32);
            asm volatile("s_wait_asynccnt 0x3" ::: "memory");   // chunk `kk` landed
        } else {
            asm volatile("s_wait_asynccnt 0x0" ::: "memory");
        }
        __syncthreads();

        const unsigned short* As  = sBuf + cur * BUFELT;        // [64][32]
        const unsigned short* Bst = As + 64 * 32;               // [128 n][32 k]

        // A frag: lanes<16 hold K 0-7 & 16-23 of row m; lanes>=16 hold K 8-15 & 24-31
        const unsigned short* ap = As + (rstrip * 16 + m16) * 32;
        v16b afrag = ldfrag(ap + hf * 8, ap + 16 + hf * 8);
#pragma unroll
        for (int ct = 0; ct < 4; ++ct) {
            // B frag: lanes<16 hold K 0-7 & 16-23 of col n; lanes>=16 hold K 8-15 & 24-31
            int n = chalf * 64 + ct * 16 + m16;
            const unsigned short* bp = Bst + n * 32 + hf * 8;
            v16b bfrag = ldfrag(bp, bp + 16);
            cacc[ct] = __builtin_amdgcn_wmma_f32_16x16x32_bf16(
                false, afrag, false, bfrag, (short)0, cacc[ct], false, false);
        }
        __syncthreads();   // release buffer `cur` for the prefetch 2 chunks ahead
        cur ^= 1;
    }

    // ---- Epilogue: C (64x128) -> bf16 LDS, then x Wt (K=128), +bt, relu ----
    unsigned short* Cs = sBuf;   // [64][128] = 8192 ushort, fits in the 12288 of sBuf
#pragma unroll
    for (int ct = 0; ct < 4; ++ct) {
        int colb = chalf * 64 + ct * 16;
#pragma unroll
        for (int j = 0; j < 8; ++j)
            Cs[(rstrip * 16 + hf * 8 + j) * 128 + colb + m16] = f2bf(cacc[ct][j]);
    }
    __syncthreads();

    v8f d[4];
#pragma unroll
    for (int i = 0; i < 4; ++i) d[i] = vzero8();
#pragma unroll
    for (int kc = 0; kc < 128; kc += 32) {
        const unsigned short* ap = Cs + (rstrip * 16 + m16) * 128 + kc;
        v16b afrag = ldfrag(ap + hf * 8, ap + 16 + hf * 8);
#pragma unroll
        for (int ct = 0; ct < 4; ++ct) {
            int n = chalf * 64 + ct * 16 + m16;
            const unsigned short* bp = sWt + n * 128 + kc + hf * 8;
            v16b bfrag = ldfrag(bp, bp + 16);
            d[ct] = __builtin_amdgcn_wmma_f32_16x16x32_bf16(
                false, afrag, false, bfrag, (short)0, d[ct], false, false);
        }
    }
#pragma unroll
    for (int ct = 0; ct < 4; ++ct) {
        int col = chalf * 64 + ct * 16 + m16;
        float btv = bt[col];
#pragma unroll
        for (int j = 0; j < 8; ++j) {
            int rg = rowBase + rstrip * 16 + hf * 8 + j;
            float v = d[ct][j] + btv;
            kout[(size_t)rg * NC + col] = v > 0.0f ? v : 0.0f;
        }
    }
}

// ---------------------------------------------------------------------------
extern "C" void kernel_launch(void* const* d_in, const int* in_sizes, int n_in,
                              void* d_out, int out_size, void* d_ws, size_t ws_size,
                              hipStream_t stream) {
    const float* vt   = (const float*)d_in[0];
    const float* y1   = (const float*)d_in[1];
    const float* y2   = (const float*)d_in[2];
    const float* A    = (const float*)d_in[3];
    const float* Wi1a = (const float*)d_in[4];
    const float* bi1a = (const float*)d_in[5];
    const float* Wi1b = (const float*)d_in[6];
    const float* bi1b = (const float*)d_in[7];
    const float* Wi2a = (const float*)d_in[8];
    const float* bi2a = (const float*)d_in[9];
    const float* Wi2b = (const float*)d_in[10];
    const float* bi2b = (const float*)d_in[11];
    const float* Wt   = (const float*)d_in[12];
    const float* bt   = (const float*)d_in[13];
    const float* Wo1  = (const float*)d_in[14];
    const float* bo1  = (const float*)d_in[15];
    const float* Wo2  = (const float*)d_in[16];
    const float* bo2  = (const float*)d_in[17];

    float* out1 = (float*)d_out;          // [20][8000]
    float* out2 = out1 + 20 * NN;         // [20][8000]
    float* x0   = out2 + 20 * NN;         // [8000][16]

    char* w = (char*)d_ws;
    unsigned short* Abf = (unsigned short*)(w);                 // 128,000,000 B
    unsigned short* WtT = (unsigned short*)(w + 128000000);     //      32,768 B
    float* s    = (float*)(w + 128032768);                      //   4,096,000 B
    float* acc  = (float*)(w + 132128768);                      //   4,096,000 B
    float* kbuf = (float*)(w + 136224768);                      //   4,096,000 B
    unsigned short* uT = (unsigned short*)(w + 140320768);      //   2,048,000 B

    convA_kernel<<<31250, 256, 0, stream>>>(A, Abf);
    convW_kernel<<<1, 256, 0, stream>>>(Wt, WtT);
    emb_kernel<<<NN, 128, 0, stream>>>(y1, y2, Wi1a, bi1a, Wi1b, bi1b,
                                       Wi2a, bi2a, Wi2b, bi2b, s, x0);
    out_kernel<<<32, 256, 0, stream>>>(s, Wo1, bo1, Wo2, bo2, out1, out2, 0);
    combine_kernel<<<63, 256, 0, stream>>>(s, s, acc, s, uT, vt, 0, 0, 0.f, 0.f);

    for (int t = 0; t < 19; ++t) {
        f_gemm<<<125, 256, 0, stream>>>(Abf, uT, WtT, bt, kbuf);                       // k1
        combine_kernel<<<63, 256, 0, stream>>>(s, kbuf, acc, s, uT, vt, t, 1, 1.f/6.f, 0.5f);
        f_gemm<<<125, 256, 0, stream>>>(Abf, uT, WtT, bt, kbuf);                       // k2
        combine_kernel<<<63, 256, 0, stream>>>(s, kbuf, acc, s, uT, vt, t, 2, 1.f/3.f, 0.5f);
        f_gemm<<<125, 256, 0, stream>>>(Abf, uT, WtT, bt, kbuf);                       // k3
        combine_kernel<<<63, 256, 0, stream>>>(s, kbuf, acc, s, uT, vt, t, 2, 1.f/3.f, 1.0f);
        f_gemm<<<125, 256, 0, stream>>>(Abf, uT, WtT, bt, kbuf);                       // k4
        combine_kernel<<<63, 256, 0, stream>>>(s, kbuf, acc, s, uT, vt, t, 3, 1.f/6.f, 0.f);
        out_kernel<<<32, 256, 0, stream>>>(s, Wo1, bo1, Wo2, bo2, out1, out2, t + 1);
    }
}
